// TopoDecoder_8418135900639
// MI455X (gfx1250) — matrix operand order repaired and checked by reading, not compiled, not run on hardware
//
#include <hip/hip_runtime.h>
#include <hip/hip_bf16.h>

typedef __bf16 bf16_t;
typedef __attribute__((ext_vector_type(16))) __bf16 v16bf;
typedef __attribute__((ext_vector_type(8)))  __bf16 v8bf;
typedef __attribute__((ext_vector_type(8)))  float  v8f;
typedef __attribute__((ext_vector_type(4)))  int    v4i_t;
typedef __attribute__((address_space(1))) v4i_t* gp4_t;
typedef __attribute__((address_space(3))) v4i_t* sp4_t;

#define B_     2
#define N_     256
#define C_     256
#define HF_    64
#define WF_    64
#define ROI_   15
#define P2_    225
#define PPAD_  240
#define HID_   256
#define HEADS_ 8
#define DH_    32
#define DEPTH_ 3
#define FFN_   1024
#define NQ_    16
#define PATCH_ 512
#define LN_EPS_ 1e-5f
#define SCALE_ 0.17677669529663687f   /* 32^-0.5 */

#if defined(__has_builtin)
#  if __has_builtin(__builtin_amdgcn_global_load_async_to_lds_b128) && \
      __has_builtin(__builtin_amdgcn_s_wait_asynccnt)
#    define USE_ASYNC_COPY 1
#  endif
#endif

// ---------------- fragment helpers (CDNA5 wave32 WMMA layouts) ----------------

__device__ __forceinline__ v8f zero8() {
  v8f z;
#pragma unroll
  for (int i = 0; i < 8; ++i) z[i] = 0.f;
  return z;
}

__device__ __forceinline__ v16bf mkfrag(v8bf lo, v8bf hi) {
  return __builtin_shufflevector(lo, hi, 0,1,2,3,4,5,6,7,8,9,10,11,12,13,14,15);
}

// A fragment 16x32 bf16 from row-major [M][K] (lane m = lane&15):
//   elems 0..7 -> K = k0 + 8*(lane>>4) + e ; elems 8..15 -> K = k0+16+8*(lane>>4)+(e-8)
__device__ __forceinline__ v16bf load_a(const bf16_t* Abase, int lda, int k0, int lane) {
  int row = lane & 15;
  int kh  = (lane >> 4) << 3;
  const bf16_t* r = Abase + row * lda + k0 + kh;
  v8bf lo = *(const v8bf*)r;
  v8bf hi = *(const v8bf*)(r + 16);
  return mkfrag(lo, hi);
}

// B fragment for X @ M^T where M is row-major bf16 [N][K] (LDS or global):
//   lane n = n0 + (lane&15); elems 0..7 -> K = k0 + 8*(lane>>4) + e ; elems 8..15 -> K = k0+16+8*(lane>>4)+(e-8)
__device__ __forceinline__ v16bf load_bT(const bf16_t* Mb, int ldm, int n0, int k0, int lane) {
  const bf16_t* r = Mb + (size_t)(n0 + (lane & 15)) * ldm + k0 + ((lane >> 4) << 3);
  v8bf lo = *(const v8bf*)r;
  v8bf hi = *(const v8bf*)(r + 16);
  return mkfrag(lo, hi);
}

__device__ __forceinline__ v8f wmma_bf(v16bf a, v16bf b, v8f c) {
  return __builtin_amdgcn_wmma_f32_16x16x32_bf16(false, a, false, b, (short)0, c, false, false);
}

// D tile stores: D elem i -> row = m0 + 8*(lane>>4) + i, col = n0 + (lane&15)
__device__ __forceinline__ void stD_f32(float* M, int ldm, int m0, int n0, v8f d, int lane) {
  int col = n0 + (lane & 15), rb = m0 + ((lane >> 4) << 3);
#pragma unroll
  for (int i = 0; i < 8; ++i) M[(rb + i) * ldm + col] = d[i];
}
__device__ __forceinline__ void stD_add_f32(float* M, int ldm, int m0, int n0, v8f d, int lane) {
  int col = n0 + (lane & 15), rb = m0 + ((lane >> 4) << 3);
#pragma unroll
  for (int i = 0; i < 8; ++i) M[(rb + i) * ldm + col] += d[i];
}
__device__ __forceinline__ void stD_bf16(bf16_t* M, int ldm, int m0, int n0, v8f d, int lane) {
  int col = n0 + (lane & 15), rb = m0 + ((lane >> 4) << 3);
#pragma unroll
  for (int i = 0; i < 8; ++i) M[(rb + i) * ldm + col] = (bf16_t)d[i];
}
__device__ __forceinline__ void stD_bf16T(bf16_t* M, int ldm, int m0, int n0, v8f d, int lane) {
  int col = n0 + (lane & 15), rb = m0 + ((lane >> 4) << 3);
#pragma unroll
  for (int i = 0; i < 8; ++i) M[(size_t)col * ldm + rb + i] = (bf16_t)d[i];
}

// generic tile: A(bf16 LDS, row-major, base already at m0) @ W^T (bf16 [N][K], L2-resident)
__device__ __forceinline__ v8f gemmW(const bf16_t* Abase, int lda,
                                     const bf16_t* W, int ldw, int n0, int K, int lane) {
  v8f acc = zero8();
  for (int k0 = 0; k0 < K; k0 += 32) {
    v16bf a = load_a(Abase, lda, k0, lane);
    v16bf b = load_bT(W, ldw, n0, k0, lane);
    acc = wmma_bf(a, b, acc);
  }
  return acc;
}

// ---------------- kernel 0: one-time f32 -> bf16 weight conversion ----------------

__global__ void k_cvt(const float* __restrict__ src, bf16_t* __restrict__ dst, int n) {
  int i = blockIdx.x * 256 + threadIdx.x;
  if (i < n) dst[i] = (bf16_t)src[i];
}

// ---------------- kernel 1: ROI bilinear sample + projection ----------------

__global__ __launch_bounds__(256, 1)
void k_roi_proj(const float* __restrict__ img, const float* __restrict__ kps,
                const bf16_t* __restrict__ Wp, const float* __restrict__ bp,
                bf16_t* __restrict__ memOut) {
  extern __shared__ __align__(16) char smem[];
  bf16_t* feat = (bf16_t*)smem;                 // [PPAD_][C_] bf16
  int seq = blockIdx.x, b = seq / N_;
  int tid = threadIdx.x, lane = tid & 31, wave = tid >> 5;

  float rcy = rintf(kps[seq * 2 + 0]);
  float rcx = rintf(kps[seq * 2 + 1]);
  const float* imgB = img + (size_t)b * C_ * HF_ * WF_;
  const float* ch = imgB + (size_t)tid * HF_ * WF_;   // channel = tid

  for (int p = 0; p < P2_; ++p) {
    int iy = p / ROI_, ix = p % ROI_;
    float gy = rcy + (float)(iy - 7);
    float gx = rcx + (float)(ix - 7);
    float gxn = gx / (float)(PATCH_ - 1) * 2.f - 1.f;
    float gyn = gy / (float)(PATCH_ - 1) * 2.f - 1.f;
    bool invalid = (gxn < -1.f) || (gyn < -1.f) || (gxn > 1.f) || (gyn > 1.f);
    if (invalid) { gxn = 0.f; gyn = 0.f; }
    float x = ((gxn + 1.f) * WF_ - 1.f) * 0.5f;
    float y = ((gyn + 1.f) * HF_ - 1.f) * 0.5f;
    float x0f = floorf(x), y0f = floorf(y);
    float wx = x - x0f, wy = y - y0f;
    int x0 = (int)x0f, y0 = (int)y0f, x1 = x0 + 1, y1 = y0 + 1;
    int xc0 = min(max(x0, 0), WF_ - 1), xc1 = min(max(x1, 0), WF_ - 1);
    int yc0 = min(max(y0, 0), HF_ - 1), yc1 = min(max(y1, 0), HF_ - 1);
    float ok00 = (y0 >= 0 && y0 < HF_ && x0 >= 0 && x0 < WF_) ? 1.f : 0.f;
    float ok01 = (y0 >= 0 && y0 < HF_ && x1 >= 0 && x1 < WF_) ? 1.f : 0.f;
    float ok10 = (y1 >= 0 && y1 < HF_ && x0 >= 0 && x0 < WF_) ? 1.f : 0.f;
    float ok11 = (y1 >= 0 && y1 < HF_ && x1 >= 0 && x1 < WF_) ? 1.f : 0.f;
    float acc = ch[yc0 * WF_ + xc0] * ((1.f - wy) * (1.f - wx) * ok00)
              + ch[yc0 * WF_ + xc1] * ((1.f - wy) * wx * ok01)
              + ch[yc1 * WF_ + xc0] * (wy * (1.f - wx) * ok10)
              + ch[yc1 * WF_ + xc1] * (wy * wx * ok11);
    if (invalid) acc = 0.f;
    feat[p * C_ + tid] = (bf16_t)acc;
  }
  for (int i = tid; i < (PPAD_ - P2_) * C_; i += 256) feat[P2_ * C_ + i] = (bf16_t)0.f;
  __syncthreads();

  // mem = feat @ proj_w^T + proj_b  (240x256 @ 256x256), 240 tiles over 8 waves
  bf16_t* dst = memOut + (size_t)seq * PPAD_ * HID_;
  for (int t = wave; t < (PPAD_ / 16) * (HID_ / 16); t += 8) {
    int m0 = (t / 16) * 16, n0 = (t % 16) * 16;
    v8f acc = gemmW(feat + m0 * C_, C_, Wp, C_, n0, C_, lane);
    float bb = bp[n0 + (lane & 15)];
#pragma unroll
    for (int i = 0; i < 8; ++i) acc[i] += bb;
    stD_bf16(dst, HID_, m0, n0, acc, lane);
  }
}

// ---------------- kernel 2: 3-layer decoder, one block per sequence ----------------

__device__ __forceinline__ void layer_norm_block(float* tgtF, bf16_t* tgtB,
                                                 const float* w, const float* b,
                                                 float* red, int tid) {
  if (tid < 16) {
    float mu = 0.f;
    for (int c = 0; c < HID_; ++c) mu += tgtF[tid * HID_ + c];
    mu *= (1.f / HID_);
    float var = 0.f;
    for (int c = 0; c < HID_; ++c) { float d = tgtF[tid * HID_ + c] - mu; var += d * d; }
    var *= (1.f / HID_);
    red[tid * 2] = mu;
    red[tid * 2 + 1] = rsqrtf(var + LN_EPS_);
  }
  __syncthreads();
  for (int i = tid; i < NQ_ * HID_; i += 256) {
    int r = i >> 8, c = i & 255;
    float v = (tgtF[i] - red[r * 2]) * red[r * 2 + 1] * w[c] + b[c];
    tgtF[i] = v;
    tgtB[i] = (bf16_t)v;
  }
  __syncthreads();
}

__global__ __launch_bounds__(256, 1)
void k_decoder(const bf16_t* __restrict__ memW, const int* __restrict__ valid,
               const float* __restrict__ qemb,
               const bf16_t* __restrict__ sqkvw, const float* __restrict__ sqkvb,
               const bf16_t* __restrict__ soutw, const float* __restrict__ soutb,
               const bf16_t* __restrict__ cqkvw, const float* __restrict__ cqkvb,
               const bf16_t* __restrict__ coutw, const float* __restrict__ coutb,
               const bf16_t* __restrict__ f1w, const float* __restrict__ f1b,
               const bf16_t* __restrict__ f2w, const float* __restrict__ f2b,
               const float* __restrict__ ln1w, const float* __restrict__ ln1b,
               const float* __restrict__ ln2w, const float* __restrict__ ln2b,
               const float* __restrict__ ln3w, const float* __restrict__ ln3b,
               const float* __restrict__ outw, const float* __restrict__ outb,
               float* __restrict__ out) {
  extern __shared__ __align__(16) char smem[];
  char* p = smem;
  bf16_t* memL = (bf16_t*)p; p += PPAD_ * HID_ * 2;     // 122880
  float*  tgtF = (float*)p;  p += NQ_ * HID_ * 4;       // 16384
  bf16_t* tgtB = (bf16_t*)p; p += NQ_ * HID_ * 2;       // 8192
  bf16_t* qB   = (bf16_t*)p; p += NQ_ * HID_ * 2;       // 8192
  bf16_t* kB   = (bf16_t*)p; p += NQ_ * HID_ * 2;       // 8192
  bf16_t* vT   = (bf16_t*)p; p += HID_ * 32 * 2;        // 16384  [feat 256][key pad 32]
  bf16_t* pB   = (bf16_t*)p; p += 4096 * 2;             // 8192   self [8][16][32] / cross [16][256]
  float*  sF   = (float*)p;  p += NQ_ * PPAD_ * 4;      // 15360
  bf16_t* Kh   = (bf16_t*)p; p += PPAD_ * DH_ * 2;      // 15360  [key 240][d 32]
  bf16_t* VhT  = (bf16_t*)p; p += DH_ * 256 * 2;        // 16384  [d 32][key pad 256]
  bf16_t* oB   = (bf16_t*)p; p += NQ_ * HID_ * 2;       // 8192
  bf16_t* hidB = (bf16_t*)p; p += NQ_ * FFN_ * 2;       // 32768
  float*  red  = (float*)p;                             // 128

  int seq = blockIdx.x;
  int tid = threadIdx.x, lane = tid & 31, wave = tid >> 5;
  bool kvalid = valid[seq] != 0;

  for (int i = tid; i < NQ_ * HID_; i += 256) tgtF[i] = qemb[i];

  // bring this sequence's mem tile (120 KB bf16) into LDS
#ifdef USE_ASYNC_COPY
  {
    char* g = (char*)(memW + (size_t)seq * PPAD_ * HID_);
    char* s = (char*)memL;
    for (int i = tid * 16; i < PPAD_ * HID_ * 2; i += 256 * 16)
      __builtin_amdgcn_global_load_async_to_lds_b128(
          (gp4_t)(g + i), (sp4_t)(s + i), 0, 0);
    __builtin_amdgcn_s_wait_asynccnt(0);
  }
#else
  {
    const uint4* s4 = (const uint4*)(memW + (size_t)seq * PPAD_ * HID_);
    uint4* d4 = (uint4*)memL;
    for (int i = tid; i < PPAD_ * HID_ * 2 / 16; i += 256) d4[i] = s4[i];
  }
#endif
  __syncthreads();

  for (int l = 0; l < DEPTH_; ++l) {
    const bf16_t* Wsq = sqkvw + (size_t)l * 768 * HID_;  const float* bsq = sqkvb + l * 768;
    const bf16_t* Wso = soutw + (size_t)l * HID_ * HID_; const float* bso = soutb + l * HID_;
    const bf16_t* Wcq = cqkvw + (size_t)l * 768 * HID_;  const float* bcq = cqkvb + l * 768;
    const bf16_t* Wco = coutw + (size_t)l * HID_ * HID_; const float* bco = coutb + l * HID_;
    const bf16_t* W1  = f1w + (size_t)l * FFN_ * HID_;   const float* b1  = f1b + l * FFN_;
    const bf16_t* W2  = f2w + (size_t)l * HID_ * FFN_;   const float* b2  = f2b + l * HID_;

    // -- stage 1: bf16 tgt, zero padded scratch regions
    for (int i = tid; i < NQ_ * HID_; i += 256) tgtB[i] = (bf16_t)tgtF[i];
    for (int i = tid; i < HID_ * 32; i += 256) vT[i] = (bf16_t)0.f;
    for (int i = tid; i < 4096; i += 256) pB[i] = (bf16_t)0.f;
    __syncthreads();

    // -- stage 2: self QKV (16x768, K=256): 48 tiles
    for (int t = wave; t < 48; t += 8) {
      int n0 = t * 16;
      v8f acc = gemmW(tgtB, HID_, Wsq, HID_, n0, HID_, lane);
      float bb = bsq[n0 + (lane & 15)];
#pragma unroll
      for (int i = 0; i < 8; ++i) acc[i] += bb;
      if (n0 < 256)      stD_bf16(qB, HID_, 0, n0, acc, lane);
      else if (n0 < 512) stD_bf16(kB, HID_, 0, n0 - 256, acc, lane);
      else               stD_bf16T(vT, 32, 0, n0 - 512, acc, lane);   // vT[feat][key]
    }
    __syncthreads();

    // -- stage 3: self scores s_h = q_h k_h^T (one 16x16x32 per head)
    {
      int h = wave;
      v16bf a  = load_a(qB + h * DH_, HID_, 0, lane);
      v16bf bb = load_bT(kB + h * DH_, HID_, 0, 0, lane);
      v8f s = wmma_bf(a, bb, zero8());
      stD_f32(sF + h * 256, 16, 0, 0, s, lane);
    }
    __syncthreads();
    // softmax (no mask), probs bf16 -> pB[h][16][32] (cols 16..31 stay zero)
    for (int t = tid; t < 128; t += 256) {
      int h = t >> 4, r = t & 15;
      float* row = sF + h * 256 + r * 16;
      float mx = -1e30f;
      for (int c = 0; c < 16; ++c) mx = fmaxf(mx, row[c] * SCALE_);
      float e[16], sum = 0.f;
      for (int c = 0; c < 16; ++c) { e[c] = __expf(row[c] * SCALE_ - mx); sum += e[c]; }
      float inv = 1.f / sum;
      bf16_t* pr = pB + (h * 16 + r) * 32;
      for (int c = 0; c < 16; ++c) pr[c] = (bf16_t)(e[c] * inv);
    }
    __syncthreads();

    // -- stage 4: o_h = p_h @ v_h : 16 tiles (head, d-tile), K=32 (upper half zero)
    for (int t = wave; t < 16; t += 8) {
      int h = t >> 1, dt = (t & 1) * 16;
      v16bf a  = load_a(pB + h * 16 * 32, 32, 0, lane);
      v16bf bb = load_bT(vT + (h * DH_) * 32, 32, dt, 0, lane);
      v8f o = wmma_bf(a, bb, zero8());
      stD_bf16(oB, HID_, 0, h * DH_ + dt, o, lane);
    }
    __syncthreads();

    // -- stage 5: self out-proj + residual
    for (int t = wave; t < 16; t += 8) {
      int n0 = t * 16;
      v8f acc = gemmW(oB, HID_, Wso, HID_, n0, HID_, lane);
      float bb = bso[n0 + (lane & 15)];
#pragma unroll
      for (int i = 0; i < 8; ++i) acc[i] += bb;
      stD_add_f32(tgtF, HID_, 0, n0, acc, lane);
    }
    __syncthreads();
    layer_norm_block(tgtF, tgtB, ln1w + l * HID_, ln1b + l * HID_, red, tid);

    // -- stage 7: cross q projection; zero VhT pad keys
    for (int t = wave; t < 16; t += 8) {
      int n0 = t * 16;
      v8f acc = gemmW(tgtB, HID_, Wcq, HID_, n0, HID_, lane);
      float bb = bcq[n0 + (lane & 15)];
#pragma unroll
      for (int i = 0; i < 8; ++i) acc[i] += bb;
      stD_bf16(qB, HID_, 0, n0, acc, lane);
    }
    for (int i = tid; i < DH_ * 16; i += 256) {
      int d = i >> 4, k = 240 + (i & 15);
      VhT[d * 256 + k] = (bf16_t)0.f;
    }
    __syncthreads();

    // -- per-head cross attention
    for (int h = 0; h < HEADS_; ++h) {
      const bf16_t* Wk = Wcq + (size_t)(256 + h * DH_) * HID_;
      const bf16_t* Wv = Wcq + (size_t)(512 + h * DH_) * HID_;
      // K_h (240x32) and V_h^T: 60 tiles over 8 waves (K=256)
      for (int t = wave; t < 60; t += 8) {
        if (t < 30) {
          int m0 = (t >> 1) * 16, n0 = (t & 1) * 16;
          v8f acc = gemmW(memL + m0 * HID_, HID_, Wk, HID_, n0, HID_, lane);
          float bb = bcq[256 + h * DH_ + n0 + (lane & 15)];
#pragma unroll
          for (int i = 0; i < 8; ++i) acc[i] += bb;
          stD_bf16(Kh, DH_, m0, n0, acc, lane);
        } else {
          int tt = t - 30;
          int m0 = (tt >> 1) * 16, n0 = (tt & 1) * 16;
          v8f acc = gemmW(memL + m0 * HID_, HID_, Wv, HID_, n0, HID_, lane);
          float bb = bcq[512 + h * DH_ + n0 + (lane & 15)];
#pragma unroll
          for (int i = 0; i < 8; ++i) acc[i] += bb;
          stD_bf16T(VhT, 256, m0, n0, acc, lane);   // VhT[d][key]
        }
      }
      __syncthreads();
      // scores 16x240: 15 tiles, K=32
      for (int t = wave; t < 15; t += 8) {
        int n0 = t * 16;
        v16bf a  = load_a(qB + h * DH_, HID_, 0, lane);
        v16bf bb = load_bT(Kh, DH_, n0, 0, lane);
        v8f s = wmma_bf(a, bb, zero8());
        stD_f32(sF, PPAD_, 0, n0, s, lane);
      }
      __syncthreads();
      // masked softmax -> pB as aB[16][256] bf16 (keys 240..255 zero)
      if (tid < 16) {
        int r = tid;
        float mx = -1e30f;
        for (int c = 0; c < P2_; ++c)
          if (kvalid || c == 0) mx = fmaxf(mx, sF[r * PPAD_ + c] * SCALE_);
        float sum = 0.f;
        for (int c = 0; c < PPAD_; ++c) {
          float v = 0.f;
          if (c < P2_ && (kvalid || c == 0)) v = __expf(sF[r * PPAD_ + c] * SCALE_ - mx);
          sF[r * PPAD_ + c] = v;
          sum += v;
        }
        float inv = 1.f / sum;
        bf16_t* pr = pB + r * 256;
        for (int c = 0; c < PPAD_; ++c) pr[c] = (bf16_t)(sF[r * PPAD_ + c] * inv);
        for (int c = PPAD_; c < 256; ++c) pr[c] = (bf16_t)0.f;
      }
      __syncthreads();
      // o_h = a @ V_h : 2 tiles, K=256
      for (int t = wave; t < 2; t += 8) {
        int dt = t * 16;
        v8f acc = zero8();
        for (int k0 = 0; k0 < 256; k0 += 32) {
          v16bf a  = load_a(pB, 256, k0, lane);
          v16bf bb = load_bT(VhT, 256, dt, k0, lane);
          acc = wmma_bf(a, bb, acc);
        }
        stD_bf16(oB, HID_, 0, h * DH_ + dt, acc, lane);
      }
      __syncthreads();
    }

    // -- stage 8: cross out-proj + residual
    for (int t = wave; t < 16; t += 8) {
      int n0 = t * 16;
      v8f acc = gemmW(oB, HID_, Wco, HID_, n0, HID_, lane);
      float bb = bco[n0 + (lane & 15)];
#pragma unroll
      for (int i = 0; i < 8; ++i) acc[i] += bb;
      stD_add_f32(tgtF, HID_, 0, n0, acc, lane);
    }
    __syncthreads();
    layer_norm_block(tgtF, tgtB, ln2w + l * HID_, ln2b + l * HID_, red, tid);

    // -- FFN1: 16x1024, K=256, relu
    for (int t = wave; t < 64; t += 8) {
      int n0 = t * 16;
      v8f acc = gemmW(tgtB, HID_, W1, HID_, n0, HID_, lane);
      float bb = b1[n0 + (lane & 15)];
#pragma unroll
      for (int i = 0; i < 8; ++i) acc[i] = fmaxf(acc[i] + bb, 0.f);
      stD_bf16(hidB, FFN_, 0, n0, acc, lane);
    }
    __syncthreads();
    // -- FFN2: 16x256, K=1024, + residual
    for (int t = wave; t < 16; t += 8) {
      int n0 = t * 16;
      v8f acc = gemmW(hidB, FFN_, W2, FFN_, n0, FFN_, lane);
      float bb = b2[n0 + (lane & 15)];
#pragma unroll
      for (int i = 0; i < 8; ++i) acc[i] += bb;
      stD_add_f32(tgtF, HID_, 0, n0, acc, lane);
    }
    __syncthreads();
    layer_norm_block(tgtF, tgtB, ln3w + l * HID_, ln3b + l * HID_, red, tid);
  }

  // output head: (16 x 4)
  if (tid < 64) {
    int q = tid >> 2, j = tid & 3;
    float acc = outb[j];
    for (int c = 0; c < HID_; ++c) acc += tgtF[q * HID_ + c] * outw[j * HID_ + c];
    out[(size_t)seq * NQ_ * 4 + q * 4 + j] = acc;
  }
}

// ---------------- launch ----------------

static constexpr size_t LDS1 = (size_t)PPAD_ * C_ * 2;                      // 122880
static constexpr size_t LDS2 = (size_t)PPAD_ * HID_ * 2 + NQ_ * HID_ * 4 +
                               4 * (NQ_ * HID_ * 2) + HID_ * 32 * 2 + 4096 * 2 +
                               NQ_ * PPAD_ * 4 + PPAD_ * DH_ * 2 + DH_ * 256 * 2 +
                               NQ_ * FFN_ * 2 + 128;                        // 276608

// bf16 weight workspace layout (elements), after the mem tensor
static constexpr size_t WS_MEM   = (size_t)B_ * N_ * PPAD_ * HID_;   // 31,457,280 elems
static constexpr size_t SZ_PROJ  = (size_t)HID_ * C_;                // 65536
static constexpr size_t SZ_SQKV  = (size_t)DEPTH_ * 3 * HID_ * HID_; // 589824
static constexpr size_t SZ_SOUT  = (size_t)DEPTH_ * HID_ * HID_;     // 196608
static constexpr size_t SZ_CQKV  = SZ_SQKV;
static constexpr size_t SZ_COUT  = SZ_SOUT;
static constexpr size_t SZ_F1    = (size_t)DEPTH_ * FFN_ * HID_;     // 786432
static constexpr size_t SZ_F2    = SZ_F1;

extern "C" void kernel_launch(void* const* d_in, const int* in_sizes, int n_in,
                              void* d_out, int out_size, void* d_ws, size_t ws_size,
                              hipStream_t stream) {
  const float* img    = (const float*)d_in[0];
  const float* kps    = (const float*)d_in[1];
  const int*   valid  = (const int*)d_in[2];
  const float* proj_w = (const float*)d_in[3];
  const float* proj_b = (const float*)d_in[4];
  const float* qemb   = (const float*)d_in[5];
  const float* sqkvw  = (const float*)d_in[6];
  const float* sqkvb  = (const float*)d_in[7];
  const float* soutw  = (const float*)d_in[8];
  const float* soutb  = (const float*)d_in[9];
  const float* cqkvw  = (const float*)d_in[10];
  const float* cqkvb  = (const float*)d_in[11];
  const float* coutw  = (const float*)d_in[12];
  const float* coutb  = (const float*)d_in[13];
  const float* f1w    = (const float*)d_in[14];
  const float* f1b    = (const float*)d_in[15];
  const float* f2w    = (const float*)d_in[16];
  const float* f2b    = (const float*)d_in[17];
  const float* ln1w   = (const float*)d_in[18];
  const float* ln1b   = (const float*)d_in[19];
  const float* ln2w   = (const float*)d_in[20];
  const float* ln2b   = (const float*)d_in[21];
  const float* ln3w   = (const float*)d_in[22];
  const float* ln3b   = (const float*)d_in[23];
  const float* outw   = (const float*)d_in[24];
  const float* outb   = (const float*)d_in[25];
  float* out = (float*)d_out;

  bf16_t* memWs = (bf16_t*)d_ws;           // ~60 MB bf16 mem tensor (L2-resident)
  bf16_t* wsp = memWs + WS_MEM;
  bf16_t* pW  = wsp;            wsp += SZ_PROJ;
  bf16_t* sqW = wsp;            wsp += SZ_SQKV;
  bf16_t* soW = wsp;            wsp += SZ_SOUT;
  bf16_t* cqW = wsp;            wsp += SZ_CQKV;
  bf16_t* coW = wsp;            wsp += SZ_COUT;
  bf16_t* f1W = wsp;            wsp += SZ_F1;
  bf16_t* f2W = wsp;            wsp += SZ_F2;

  // one-time (per call) f32 -> bf16 weight conversion; all L2-resident
  k_cvt<<<(int)((SZ_PROJ + 255) / 256), 256, 0, stream>>>(proj_w, pW, (int)SZ_PROJ);
  k_cvt<<<(int)((SZ_SQKV + 255) / 256), 256, 0, stream>>>(sqkvw, sqW, (int)SZ_SQKV);
  k_cvt<<<(int)((SZ_SOUT + 255) / 256), 256, 0, stream>>>(soutw, soW, (int)SZ_SOUT);
  k_cvt<<<(int)((SZ_CQKV + 255) / 256), 256, 0, stream>>>(cqkvw, cqW, (int)SZ_CQKV);
  k_cvt<<<(int)((SZ_COUT + 255) / 256), 256, 0, stream>>>(coutw, coW, (int)SZ_COUT);
  k_cvt<<<(int)((SZ_F1   + 255) / 256), 256, 0, stream>>>(f1w, f1W, (int)SZ_F1);
  k_cvt<<<(int)((SZ_F2   + 255) / 256), 256, 0, stream>>>(f2w, f2W, (int)SZ_F2);

  (void)hipFuncSetAttribute((const void*)k_roi_proj,
                            hipFuncAttributeMaxDynamicSharedMemorySize, (int)LDS1);
  (void)hipFuncSetAttribute((const void*)k_decoder,
                            hipFuncAttributeMaxDynamicSharedMemorySize, (int)LDS2);

  k_roi_proj<<<B_ * N_, 256, LDS1, stream>>>(img, kps, pW, proj_b, memWs);
  k_decoder<<<B_ * N_, 256, LDS2, stream>>>(memWs, valid, qemb,
      sqW, sqkvb, soW, soutb, cqW, cqkvb, coW, coutb,
      f1W, f1b, f2W, f2b, ln1w, ln1b, ln2w, ln2b, ln3w, ln3b, outw, outb, out);
}
